// Graph_MERFISH_31542239822523
// MI455X (gfx1250) — matrix operand order repaired and checked by reading, not compiled, not run on hardware
//
#include <hip/hip_runtime.h>
#include <hip/hip_bf16.h>
#include <math.h>

typedef __attribute__((ext_vector_type(16))) __bf16        v16bf;
typedef __attribute__((ext_vector_type(8)))  float         v8f;
typedef __attribute__((ext_vector_type(4)))  unsigned int  v4u;

union Frag16 { v16bf v; unsigned int u[8]; };

__device__ __forceinline__ unsigned short f2bf(float f) {
    unsigned int u = __float_as_uint(f);
    u += 0x7FFFu + ((u >> 16) & 1u);   // round-to-nearest-even
    return (unsigned short)(u >> 16);
}
__device__ __forceinline__ unsigned int pack2bf(float a, float b) {
    return (unsigned int)f2bf(a) | ((unsigned int)f2bf(b) << 16);
}

// ---------------------------------------------------------------------------
// GEMM: C[M,N] = A[M,K] @ B[K,N] (+bias) (+relu), f32 in/out, bf16 WMMA core.
// Block = 256 threads = 8 waves. Block tile 128(M) x 64(N), K-step 32.
// LDS tiles are stored pre-packed in WMMA fragment order (dword = bf16 pair)
// with a 20-dword row pitch: 16B-aligned, conflict-free ds_load_b128 reads.
// Double-buffered so tile k+1 global loads overlap tile k WMMAs.
// Full M-tiles take a uniform unguarded load/store path (no exec masking).
// ---------------------------------------------------------------------------
#define TPITCH 20   // dwords per packed row (16 data + 4 pad)

__global__ __launch_bounds__(256) void gemm_bf16_wmma(
    const float* __restrict__ A, const float* __restrict__ B,
    float* __restrict__ C, const float* __restrict__ bias,
    int M, int K, int N, int relu)
{
    __shared__ v4u As4[2][128 * (TPITCH / 4)];   // 2 x 10 KB
    __shared__ v4u Bs4[2][64  * (TPITCH / 4)];   // 2 x 5 KB

    const int tid    = threadIdx.x;
    const int w      = tid >> 5;
    const int lane   = tid & 31;
    const int half   = lane >> 4;
    const int l16    = lane & 15;
    const int blockM = blockIdx.y * 128;
    const int blockN = blockIdx.x * 64;
    const bool fullM = (blockM + 128 <= M);      // uniform per block

    v8f acc[4] = {};
    const int ksteps = K >> 5;

    // ---- tile loader: global f32 -> packed bf16-pair dwords in LDS ----
    auto load_tile = [&](int kt, int buf) {
        unsigned int* As = (unsigned int*)As4[buf];
        unsigned int* Bs = (unsigned int*)Bs4[buf];
        const int k0 = kt << 5;
        // A tile: 128 rows x 16 kpairs, 8 dwords per thread, float2 reads
        if (fullM) {
            #pragma unroll
            for (int i = 0; i < 8; ++i) {
                int idx = tid + i * 256;
                int row = idx >> 4, kp = idx & 15;
                float2 fv = *(const float2*)&A[(size_t)(blockM + row) * K + k0 + 2 * kp];
                As[row * TPITCH + kp] = pack2bf(fv.x, fv.y);
            }
        } else {
            #pragma unroll
            for (int i = 0; i < 8; ++i) {
                int idx = tid + i * 256;
                int row = idx >> 4, kp = idx & 15;
                int gr = blockM + row;
                float2 fv = make_float2(0.0f, 0.0f);
                if (gr < M) fv = *(const float2*)&A[(size_t)gr * K + k0 + 2 * kp];
                As[row * TPITCH + kp] = pack2bf(fv.x, fv.y);
            }
        }
        // B tile: transpose-pack into Bs[col][kpair], 4 dwords per thread
        #pragma unroll
        for (int i = 0; i < 4; ++i) {
            int idx = tid + i * 256;
            int kp = idx >> 6, col = idx & 63;
            float b0 = B[(size_t)(k0 + 2 * kp) * N + blockN + col];
            float b1 = B[(size_t)(k0 + 2 * kp + 1) * N + blockN + col];
            Bs[col * TPITCH + kp] = pack2bf(b0, b1);
        }
    };

    load_tile(0, 0);

    for (int kt = 0; kt < ksteps; ++kt) {
        __syncthreads();
        if (kt + 1 < ksteps) load_tile(kt + 1, (kt + 1) & 1);

        const unsigned int* As = (const unsigned int*)As4[kt & 1];
        const unsigned int* Bs = (const unsigned int*)Bs4[kt & 1];

        // A fragment: two contiguous 16B runs per lane (ds_load_b128 x2)
        Frag16 af;
        const unsigned int* ap = As + ((w << 4) + l16) * TPITCH;
        *(v4u*)&af.u[0] = *(const v4u*)(ap + (half << 2));
        *(v4u*)&af.u[4] = *(const v4u*)(ap + 8 + (half << 2));

        #pragma unroll
        for (int nt = 0; nt < 4; ++nt) {
            // B fragment: two contiguous 16B runs per lane (ds_load_b128 x2)
            Frag16 bfm;
            const unsigned int* bp = Bs + ((nt << 4) + l16) * TPITCH;
            *(v4u*)&bfm.u[0] = *(const v4u*)(bp + (half << 2));
            *(v4u*)&bfm.u[4] = *(const v4u*)(bp + 8 + (half << 2));
            acc[nt] = __builtin_amdgcn_wmma_f32_16x16x32_bf16(
                false, af.v, false, bfm.v, (short)0, acc[nt], false, false);
        }
    }

    // Store: D element (vgpr q, lane L): M = q + 8*(L/16), N = L%16
    if (fullM) {
        #pragma unroll
        for (int nt = 0; nt < 4; ++nt) {
            int ccol = blockN + (nt << 4) + l16;
            float bv = bias ? bias[ccol] : 0.0f;
            #pragma unroll
            for (int q = 0; q < 8; ++q) {
                int r = blockM + (w << 4) + (half << 3) + q;
                float val = acc[nt][q] + bv;
                if (relu) val = fmaxf(val, 0.0f);
                C[(size_t)r * N + ccol] = val;
            }
        }
    } else {
        #pragma unroll
        for (int nt = 0; nt < 4; ++nt) {
            int ccol = blockN + (nt << 4) + l16;
            float bv = bias ? bias[ccol] : 0.0f;
            #pragma unroll
            for (int q = 0; q < 8; ++q) {
                int r = blockM + (w << 4) + (half << 3) + q;
                if (r < M) {
                    float val = acc[nt][q] + bv;
                    if (relu) val = fmaxf(val, 0.0f);
                    C[(size_t)r * N + ccol] = val;
                }
            }
        }
    }
}

// ---------------------------------------------------------------------------
// Attention scores: one block per node, wave w = head, lanes cover 64 chans.
// ---------------------------------------------------------------------------
__global__ __launch_bounds__(256) void score_kernel(
    const float* __restrict__ h, const float* __restrict__ a_src,
    const float* __restrict__ a_dst, float* __restrict__ ssrc,
    float* __restrict__ sdst)
{
    int node = blockIdx.x;
    int w = threadIdx.x >> 5;
    int lane = threadIdx.x & 31;
    const float* hr = h + (size_t)node * 512 + w * 64;
    float x0 = hr[lane], x1 = hr[lane + 32];
    float ss = x0 * a_src[w * 64 + lane] + x1 * a_src[w * 64 + lane + 32];
    float sd = x0 * a_dst[w * 64 + lane] + x1 * a_dst[w * 64 + lane + 32];
    #pragma unroll
    for (int off = 16; off; off >>= 1) {
        ss += __shfl_xor(ss, off, 32);
        sd += __shfl_xor(sd, off, 32);
    }
    if (lane == 0) { ssrc[node * 8 + w] = ss; sdst[node * 8 + w] = sd; }
}

__global__ void fill_zero(float* __restrict__ p, long long n) {
    long long i = (long long)blockIdx.x * 256 + threadIdx.x;
    if (i < n) p[i] = 0.0f;
}

__device__ __forceinline__ void edge_sd(const int* __restrict__ ei, int et,
                                        int E, int& s, int& d) {
    if (et < E) { s = ei[et]; d = ei[E + et]; }
    else        { s = et - E; d = s; }        // self loops appended
}

// leaky(0.2) logits + segment max via ordered-uint atomicMax
__global__ void edge_logits_max(const int* __restrict__ ei,
                                const float* __restrict__ ssrc,
                                const float* __restrict__ sdst,
                                float* __restrict__ ebuf,
                                unsigned int* __restrict__ emax, int E, int Et)
{
    int idx = blockIdx.x * 256 + threadIdx.x;
    if (idx >= Et * 8) return;
    int et = idx >> 3, hh = idx & 7;
    int s, d; edge_sd(ei, et, E, s, d);
    float e = ssrc[s * 8 + hh] + sdst[d * 8 + hh];
    e = e >= 0.0f ? e : 0.2f * e;
    ebuf[idx] = e;
    unsigned int u = __float_as_uint(e);
    u = (u & 0x80000000u) ? ~u : (u | 0x80000000u);
    atomicMax(&emax[d * 8 + hh], u);
}

__global__ void edge_exp_sum(const int* __restrict__ ei,
                             float* __restrict__ ebuf,
                             const unsigned int* __restrict__ emax,
                             float* __restrict__ denom, int E, int Et)
{
    int idx = blockIdx.x * 256 + threadIdx.x;
    if (idx >= Et * 8) return;
    int et = idx >> 3, hh = idx & 7;
    int s, d; edge_sd(ei, et, E, s, d);
    unsigned int u = emax[d * 8 + hh];
    float m = __uint_as_float((u & 0x80000000u) ? (u & 0x7FFFFFFFu) : ~u);
    float ex = __expf(ebuf[idx] - m);
    ebuf[idx] = ex;
    atomicAdd(&denom[d * 8 + hh], ex);
}

__global__ void edge_alpha(const int* __restrict__ ei,
                           float* __restrict__ ebuf,
                           const float* __restrict__ denom,
                           float* __restrict__ alpha_out, int E, int Et)
{
    int idx = blockIdx.x * 256 + threadIdx.x;
    if (idx >= Et * 8) return;
    int et = idx >> 3, hh = idx & 7;
    int s, d; edge_sd(ei, et, E, s, d);
    float al = ebuf[idx] / denom[d * 8 + hh];
    ebuf[idx] = al;
    alpha_out[idx] = al;
}

// one thread per (edge, channel): coalesced gather + coalesced f32 atomics
__global__ void edge_aggregate(const int* __restrict__ ei,
                               const float* __restrict__ h,
                               const float* __restrict__ ebuf,
                               float* __restrict__ agg, int E, int Et)
{
    long long idx = (long long)blockIdx.x * 256 + threadIdx.x;
    if (idx >= (long long)Et * 512) return;
    int et = (int)(idx >> 9);
    int c  = (int)(idx & 511);
    int hh = c >> 6;
    int s, d; edge_sd(ei, et, E, s, d);
    float al = ebuf[(size_t)et * 8 + hh];
    atomicAdd(&agg[(size_t)d * 512 + c], h[(size_t)s * 512 + c] * al);
}

// bias + leaky(0.01) + layernorm (+optional sigmoid). Block 256 per node.
__global__ __launch_bounds__(256) void post_kernel(
    const float* __restrict__ buf, const float* __restrict__ bias,
    const float* __restrict__ g, const float* __restrict__ beta,
    float* __restrict__ outp, int mode)
{
    __shared__ float s1[256];
    __shared__ float s2[256];
    int node = blockIdx.x;
    int t = threadIdx.x;
    const float* row = buf + (size_t)node * 512;
    float v0 = row[t] + bias[t];
    float v1 = row[t + 256] + bias[t + 256];
    v0 = v0 >= 0.0f ? v0 : 0.01f * v0;
    v1 = v1 >= 0.0f ? v1 : 0.01f * v1;
    s1[t] = v0 + v1;
    s2[t] = v0 * v0 + v1 * v1;
    __syncthreads();
    for (int off = 128; off; off >>= 1) {
        if (t < off) { s1[t] += s1[t + off]; s2[t] += s2[t + off]; }
        __syncthreads();
    }
    float mean = s1[0] * (1.0f / 512.0f);
    float var  = s2[0] * (1.0f / 512.0f) - mean * mean;
    float rstd = rsqrtf(var + 1e-5f);
    float o0 = (v0 - mean) * rstd * g[t] + beta[t];
    float o1 = (v1 - mean) * rstd * g[t + 256] + beta[t + 256];
    if (mode == 1) {
        o0 = 1.0f / (1.0f + __expf(-o0));
        o1 = 1.0f / (1.0f + __expf(-o1));
    }
    float* orow = outp + (size_t)node * 512;
    orow[t] = o0;
    orow[t + 256] = o1;
}

extern "C" void kernel_launch(void* const* d_in, const int* in_sizes, int n_in,
                              void* d_out, int out_size, void* d_ws, size_t ws_size,
                              hipStream_t stream) {
    (void)n_in; (void)out_size; (void)ws_size;
    const float* x      = (const float*)d_in[0];
    const int*   ei     = (const int*)d_in[1];
    const float* W1     = (const float*)d_in[2];
    const float* a_src1 = (const float*)d_in[3];
    const float* a_dst1 = (const float*)d_in[4];
    const float* b1     = (const float*)d_in[5];
    const float* g1     = (const float*)d_in[6];
    const float* beta1  = (const float*)d_in[7];
    const float* W2     = (const float*)d_in[8];
    const float* a_src2 = (const float*)d_in[9];
    const float* a_dst2 = (const float*)d_in[10];
    const float* b2     = (const float*)d_in[11];
    const float* g2     = (const float*)d_in[12];
    const float* beta2  = (const float*)d_in[13];
    const float* dw1    = (const float*)d_in[14];
    const float* db1    = (const float*)d_in[15];
    const float* dw2    = (const float*)d_in[16];
    const float* db2    = (const float*)d_in[17];

    const int Nn = in_sizes[0] / 512;   // 100000
    const int E  = in_sizes[1] / 2;     // 600000
    const int Et = E + Nn;              // 700000

    float* ws_f = (float*)d_ws;
    float* bufA  = ws_f;                              // N*512
    float* bufB  = ws_f + (size_t)Nn * 512;           // N*512
    float* small = ws_f + 2 * (size_t)Nn * 512;
    float* ssrc  = small;                             // N*8
    float* sdst  = small + (size_t)Nn * 8;            // N*8
    unsigned int* emax = (unsigned int*)(small + 2 * (size_t)Nn * 8);  // N*8
    float* denom = small + 3 * (size_t)Nn * 8;        // N*8
    float* ebuf  = small + 4 * (size_t)Nn * 8;        // Et*8
    float* hidden = ws_f;                             // reuse bufA+bufB: N*1024

    float* out0   = (float*)d_out;
    float* rec    = out0 + (size_t)Nn * 512;
    float* alpha1 = rec + (size_t)Nn * 512;
    float* alpha2 = alpha1 + (size_t)Et * 8;

    dim3 blk(256);
    auto cdiv = [](long long a, long long b) { return (unsigned)((a + b - 1) / b); };
    const unsigned mtiles = cdiv(Nn, 128);
    const long long eth = (long long)Et * 8;
    const long long etc = (long long)Et * 512;

    // ---- Layer 1 ----
    gemm_bf16_wmma<<<dim3(512 / 64, mtiles), blk, 0, stream>>>(
        x, W1, bufA, nullptr, Nn, 512, 512, 0);
    score_kernel<<<Nn, blk, 0, stream>>>(bufA, a_src1, a_dst1, ssrc, sdst);
    fill_zero<<<cdiv((long long)Nn * 512, 256), blk, 0, stream>>>(bufB, (long long)Nn * 512);
    fill_zero<<<cdiv((long long)Nn * 16, 256), blk, 0, stream>>>((float*)emax, (long long)Nn * 16);
    edge_logits_max<<<cdiv(eth, 256), blk, 0, stream>>>(ei, ssrc, sdst, ebuf, emax, E, Et);
    edge_exp_sum<<<cdiv(eth, 256), blk, 0, stream>>>(ei, ebuf, emax, denom, E, Et);
    edge_alpha<<<cdiv(eth, 256), blk, 0, stream>>>(ei, ebuf, denom, alpha1, E, Et);
    edge_aggregate<<<cdiv(etc, 256), blk, 0, stream>>>(ei, bufA, ebuf, bufB, E, Et);
    post_kernel<<<Nn, blk, 0, stream>>>(bufB, b1, g1, beta1, bufB, 0);

    // ---- Layer 2 ----
    gemm_bf16_wmma<<<dim3(512 / 64, mtiles), blk, 0, stream>>>(
        bufB, W2, bufA, nullptr, Nn, 512, 512, 0);
    score_kernel<<<Nn, blk, 0, stream>>>(bufA, a_src2, a_dst2, ssrc, sdst);
    fill_zero<<<cdiv((long long)Nn * 512, 256), blk, 0, stream>>>(bufB, (long long)Nn * 512);
    fill_zero<<<cdiv((long long)Nn * 16, 256), blk, 0, stream>>>((float*)emax, (long long)Nn * 16);
    edge_logits_max<<<cdiv(eth, 256), blk, 0, stream>>>(ei, ssrc, sdst, ebuf, emax, E, Et);
    edge_exp_sum<<<cdiv(eth, 256), blk, 0, stream>>>(ei, ebuf, emax, denom, E, Et);
    edge_alpha<<<cdiv(eth, 256), blk, 0, stream>>>(ei, ebuf, denom, alpha2, E, Et);
    edge_aggregate<<<cdiv(etc, 256), blk, 0, stream>>>(ei, bufA, ebuf, bufB, E, Et);
    post_kernel<<<Nn, blk, 0, stream>>>(bufB, b2, g2, beta2, out0, 1);  // sigmoid -> d_out

    // ---- Decoder ----
    gemm_bf16_wmma<<<dim3(1024 / 64, mtiles), blk, 0, stream>>>(
        out0, dw1, hidden, db1, Nn, 512, 1024, 1);   // relu epilogue
    gemm_bf16_wmma<<<dim3(512 / 64, mtiles), blk, 0, stream>>>(
        hidden, dw2, rec, db2, Nn, 1024, 512, 0);
}
